// GRU_Encoder_8349416423906
// MI455X (gfx1250) — matrix operand order repaired and checked by reading, compiled, not run on hardware
//
#include <hip/hip_runtime.h>
#include <hip/hip_bf16.h>

// ---------------- problem constants ----------------
#define B_   128
#define S_   1024
#define I_   256
#define H_   512
#define G3_  1536            // 3*H
#define NB_  64              // persistent workgroups (4 wave32 each) -> 256 tile-waves

// padded LDS leading dims (bf16 elems) to avoid 64-bank conflicts: stride mod 64 dwords = 4
#define P0_  264             // for K=256 rows
#define P1_  520             // for K=512 rows
#define LDS_ELEMS (48 * P0_ + 3 * 48 * P1_)
#define LDS_BYTES (LDS_ELEMS * 2)

typedef __attribute__((ext_vector_type(16))) __bf16 v16bf;
typedef __attribute__((ext_vector_type(8)))  __bf16 v8bf;
typedef __attribute__((ext_vector_type(8)))  float  v8f;

// ---------------- WMMA helper ----------------
static __device__ __forceinline__ v8f wmma_bf16(v16bf a, v16bf b, v8f c) {
  return __builtin_amdgcn_wmma_f32_16x16x32_bf16(false, a, false, b, (short)0, c, false, false);
}

// A-fragment (16x32, 16-bit) per ISA: lane L: row row0+(L&15);
// lanes 0-15 hold K {k0..k0+7, k0+16..k0+23}, lanes 16-31 hold K {k0+8..15, k0+24..31}.
static __device__ __forceinline__ v16bf frag_a_bf16(const __bf16* base, int ld, int row0, int k0, int lane) {
  int m  = row0 + (lane & 15);
  int kk = k0 + ((lane >> 4) << 3);
  const __bf16* p = base + (size_t)m * (size_t)ld + kk;
  v8bf lo = *(const v8bf*)(p);
  v8bf hi = *(const v8bf*)(p + 16);
  v16bf f;
#pragma unroll
  for (int j = 0; j < 8; ++j) { f[j] = lo[j]; f[j + 8] = hi[j]; }
  return f;
}

static __device__ __forceinline__ v16bf frag_a_f32(const float* base, int ld, int row0, int k0, int lane) {
  int m  = row0 + (lane & 15);
  int kk = k0 + ((lane >> 4) << 3);
  const float* p = base + (size_t)m * (size_t)ld + kk;
  v8f lo = *(const v8f*)(p);
  v8f hi = *(const v8f*)(p + 16);
  v16bf f;
#pragma unroll
  for (int j = 0; j < 8; ++j) { f[j] = (__bf16)lo[j]; f[j + 8] = (__bf16)hi[j]; }
  return f;
}

// B-fragment (32x16, 16-bit) from an LDS weight slice [48 rows x ld], row = gate*16 + col.
// lane L: column gateBase+(L&15); lanes 0-15 hold K {k0..k0+7, k0+16..k0+23}, lanes 16-31 hold K {k0+8..15, k0+24..31}.
static __device__ __forceinline__ v16bf frag_b_lds(const __bf16* Ws, int ld, int gateBase, int k0, int lane) {
  int n  = gateBase + (lane & 15);
  int kk = k0 + ((lane >> 4) << 3);
  const __bf16* p = Ws + n * ld + kk;
  v8bf lo = *(const v8bf*)(p);
  v8bf hi = *(const v8bf*)(p + 16);
  v16bf f;
#pragma unroll
  for (int j = 0; j < 8; ++j) { f[j] = lo[j]; f[j + 8] = hi[j]; }
  return f;
}

static __device__ __forceinline__ float sigm(float x) { return 1.f / (1.f + __expf(-x)); }
static __device__ __forceinline__ float tanh_f(float x) {
  float xc = fminf(fmaxf(x, -15.f), 15.f);
  float e  = __expf(2.f * xc);
  return (e - 1.f) / (e + 1.f);
}

// ---------------- grid-wide barrier (epoch counting, 64 arrivals) ----------------
static __device__ __forceinline__ void grid_sync(unsigned* bar, unsigned target) {
  __threadfence();                                   // release h-state writes
  if (threadIdx.x == 0) {
    __atomic_fetch_add(bar, 1u, __ATOMIC_ACQ_REL);
    while (__atomic_load_n(bar, __ATOMIC_ACQUIRE) < target) {
      __builtin_amdgcn_s_sleep(2);
    }
  }
  __syncthreads();
  __threadfence();                                   // acquire
}

// ---------------- one GRU-cell 16x16 tile task ----------------
static __device__ __forceinline__ void gru_layer_tile(
    const float* xF, const __bf16* xB, int ldx, int Kx,   // activation input (one of xF/xB)
    const __bf16* sWih, int ldWih,                         // LDS slices [48 x ld]
    const __bf16* sWhh, int ldWhh,
    const __bf16* hOldB, const float* hOldF,
    float bir, float biz, float bin, float bhr, float bhz, float bhn,
    float* hNewF, __bf16* hNewB,
    float* outBase, size_t outStride,
    float* hFin,
    int mtile, int col, int lane)
{
  const int m0 = mtile * 16;
  v8f gr = {0,0,0,0,0,0,0,0}, gz = gr, gn = gr;
  v8f hr = gr, hz = gr, hn = gr;

  for (int k = 0; k < Kx; k += 32) {
    v16bf a = xB ? frag_a_bf16(xB, ldx, m0, k, lane)
                 : frag_a_f32(xF, ldx, m0, k, lane);
    gr = wmma_bf16(a, frag_b_lds(sWih, ldWih, 0,  k, lane), gr);
    gz = wmma_bf16(a, frag_b_lds(sWih, ldWih, 16, k, lane), gz);
    gn = wmma_bf16(a, frag_b_lds(sWih, ldWih, 32, k, lane), gn);
  }
  for (int k = 0; k < H_; k += 32) {
    v16bf a = frag_a_bf16(hOldB, H_, m0, k, lane);
    hr = wmma_bf16(a, frag_b_lds(sWhh, ldWhh, 0,  k, lane), hr);
    hz = wmma_bf16(a, frag_b_lds(sWhh, ldWhh, 16, k, lane), hz);
    hn = wmma_bf16(a, frag_b_lds(sWhh, ldWhh, 32, k, lane), hn);
  }

  // C/D layout: VGPR r, lane L -> row m0 + r + 8*(L>>4), col (L&15)
  const int radd = (lane >> 4) << 3;
#pragma unroll
  for (int r = 0; r < 8; ++r) {
    int b = m0 + radd + r;
    float hold = hOldF[b * H_ + col];
    float rv = sigm(gr[r] + bir + hr[r] + bhr);
    float zv = sigm(gz[r] + biz + hz[r] + bhz);
    float nv = tanh_f(gn[r] + bin + rv * (hn[r] + bhn));
    float hnew = (1.f - zv) * nv + zv * hold;
    hNewF[b * H_ + col] = hnew;
    hNewB[b * H_ + col] = (__bf16)hnew;
    if (outBase) outBase[(size_t)b * outStride + col] = hnew;
    if (hFin)    hFin[b * H_ + col] = hnew;
  }
}

// ---------------- prep: weights -> bf16, zero state, reset barrier ----------------
__global__ void gru_prep(const float* __restrict__ wih0, const float* __restrict__ whh0,
                         const float* __restrict__ wih1, const float* __restrict__ whh1,
                         __bf16* __restrict__ wih0b, __bf16* __restrict__ whh0b,
                         __bf16* __restrict__ wih1b, __bf16* __restrict__ whh1b,
                         float* __restrict__ h0f, float* __restrict__ h1f,
                         __bf16* __restrict__ h0b, __bf16* __restrict__ h1b,
                         unsigned* __restrict__ bar)
{
  int i = blockIdx.x * blockDim.x + threadIdx.x;
  int stride = gridDim.x * blockDim.x;
  const int NW0 = G3_ * I_;
  const int NW  = G3_ * H_;
  const int NH  = 2 * B_ * H_;
  for (int j = i; j < NW0; j += stride) wih0b[j] = (__bf16)wih0[j];
  for (int j = i; j < NW;  j += stride) {
    whh0b[j] = (__bf16)whh0[j];
    wih1b[j] = (__bf16)wih1[j];
    whh1b[j] = (__bf16)whh1[j];
  }
  for (int j = i; j < NH; j += stride) {
    h0f[j] = 0.f; h1f[j] = 0.f;
    h0b[j] = (__bf16)0.f; h1b[j] = (__bf16)0.f;
  }
  if (i == 0) *bar = 0u;
}

// ---------------- persistent GRU kernel ----------------
__global__ void __launch_bounds__(128) gru_persistent(
    const float* __restrict__ input,
    const float* __restrict__ bih0, const float* __restrict__ bhh0,
    const float* __restrict__ bih1, const float* __restrict__ bhh1,
    const __bf16* __restrict__ wih0, const __bf16* __restrict__ whh0,
    const __bf16* __restrict__ wih1, const __bf16* __restrict__ whh1,
    float* __restrict__ h0f, float* __restrict__ h1f,
    __bf16* __restrict__ h0b, __bf16* __restrict__ h1b,
    float* __restrict__ out, float* __restrict__ h1fin,
    unsigned* __restrict__ bar)
{
  extern __shared__ __bf16 lds[];
  const int wave  = threadIdx.x >> 5;
  const int lane  = threadIdx.x & 31;
  const int ntile = blockIdx.x >> 1;                     // 0..31; waves in a block share ntile
  const int mtile = ((blockIdx.x & 1) << 2) + wave;      // 0..7
  const int HBUF  = B_ * H_;

  __bf16* sWih0 = lds;                                   // [48][P0_]
  __bf16* sWhh0 = sWih0 + 48 * P0_;                      // [48][P1_]
  __bf16* sWih1 = sWhh0 + 48 * P1_;
  __bf16* sWhh1 = sWih1 + 48 * P1_;

  // one-time LDS fill: 48 gate-rows of each weight matrix for this ntile
  for (int idx = threadIdx.x; idx < 48 * 32; idx += 128) {     // K=256 -> 32 chunks of 8
    int r = idx >> 5, q = idx & 31;
    int grow = (r >> 4) * H_ + ntile * 16 + (r & 15);
    *(v8bf*)(sWih0 + r * P0_ + q * 8) = *(const v8bf*)(wih0 + (size_t)grow * I_ + q * 8);
  }
  for (int idx = threadIdx.x; idx < 48 * 64; idx += 128) {     // K=512 -> 64 chunks of 8
    int r = idx >> 6, q = idx & 63;
    int grow = (r >> 4) * H_ + ntile * 16 + (r & 15);
    *(v8bf*)(sWhh0 + r * P1_ + q * 8) = *(const v8bf*)(whh0 + (size_t)grow * H_ + q * 8);
    *(v8bf*)(sWih1 + r * P1_ + q * 8) = *(const v8bf*)(wih1 + (size_t)grow * H_ + q * 8);
    *(v8bf*)(sWhh1 + r * P1_ + q * 8) = *(const v8bf*)(whh1 + (size_t)grow * H_ + q * 8);
  }
  __syncthreads();

  // hoist the 12 per-column bias scalars out of the time loop
  const int col = ntile * 16 + (lane & 15);
  const float bir0 = bih0[col], biz0 = bih0[col + H_], bin0 = bih0[col + 2 * H_];
  const float bhr0 = bhh0[col], bhz0 = bhh0[col + H_], bhn0 = bhh0[col + 2 * H_];
  const float bir1 = bih1[col], biz1 = bih1[col + H_], bin1 = bih1[col + 2 * H_];
  const float bhr1 = bhh1[col], bhz1 = bhh1[col + H_], bhn1 = bhh1[col + 2 * H_];

  for (int t = 0; t < S_; ++t) {
    const int oldp = t & 1, newp = oldp ^ 1;

    // ---- layer 0: x_t (fp32, row stride S*I) + h0_old -> h0_new ----
    gru_layer_tile(input + (size_t)t * I_, (const __bf16*)nullptr, S_ * I_, I_,
                   sWih0, P0_, sWhh0, P1_,
                   h0b + oldp * HBUF, h0f + oldp * HBUF,
                   bir0, biz0, bin0, bhr0, bhz0, bhn0,
                   h0f + newp * HBUF, h0b + newp * HBUF,
                   (float*)nullptr, 0, (float*)nullptr,
                   mtile, col, lane);

    grid_sync(bar, (unsigned)NB_ * (unsigned)(t + 1));   // single barrier per step

    // ---- layer 1: h0_new + h1_old -> h1_new (writes out[:,t,:]) ----
    gru_layer_tile((const float*)nullptr, h0b + newp * HBUF, H_, H_,
                   sWih1, P1_, sWhh1, P1_,
                   h1b + oldp * HBUF, h1f + oldp * HBUF,
                   bir1, biz1, bin1, bhr1, bhz1, bhn1,
                   h1f + newp * HBUF, h1b + newp * HBUF,
                   out + (size_t)t * H_, (size_t)S_ * H_,
                   (t == S_ - 1) ? h1fin : (float*)nullptr,
                   mtile, col, lane);
    // step->step hazards are covered by the next iteration's barrier
  }
}

// ---------------- launch ----------------
extern "C" void kernel_launch(void* const* d_in, const int* in_sizes, int n_in,
                              void* d_out, int out_size, void* d_ws, size_t ws_size,
                              hipStream_t stream) {
  const float* input = (const float*)d_in[0];
  const float* w_ih0 = (const float*)d_in[1];
  const float* w_hh0 = (const float*)d_in[2];
  const float* b_ih0 = (const float*)d_in[3];
  const float* b_hh0 = (const float*)d_in[4];
  const float* w_ih1 = (const float*)d_in[5];
  const float* w_hh1 = (const float*)d_in[6];
  const float* b_ih1 = (const float*)d_in[7];
  const float* b_hh1 = (const float*)d_in[8];

  float* out   = (float*)d_out;                       // [B,S,H]
  float* h1fin = out + (size_t)B_ * S_ * H_;          // final h1 [B,H]

  char* ws = (char*)d_ws;
  size_t o = 0;
  __bf16* wih0b = (__bf16*)(ws + o); o += (size_t)G3_ * I_ * 2;
  __bf16* whh0b = (__bf16*)(ws + o); o += (size_t)G3_ * H_ * 2;
  __bf16* wih1b = (__bf16*)(ws + o); o += (size_t)G3_ * H_ * 2;
  __bf16* whh1b = (__bf16*)(ws + o); o += (size_t)G3_ * H_ * 2;
  float*  h0f   = (float*)(ws + o);  o += (size_t)2 * B_ * H_ * 4;
  float*  h1f   = (float*)(ws + o);  o += (size_t)2 * B_ * H_ * 4;
  __bf16* h0b   = (__bf16*)(ws + o); o += (size_t)2 * B_ * H_ * 2;
  __bf16* h1b   = (__bf16*)(ws + o); o += (size_t)2 * B_ * H_ * 2;
  unsigned* bar = (unsigned*)(ws + o); o += 64;
  (void)ws_size; (void)in_sizes; (void)n_in; (void)out_size;

  gru_prep<<<512, 256, 0, stream>>>(w_ih0, w_hh0, w_ih1, w_hh1,
                                    wih0b, whh0b, wih1b, whh1b,
                                    h0f, h1f, h0b, h1b, bar);

  gru_persistent<<<NB_, 128, LDS_BYTES, stream>>>(input,
                                                  b_ih0, b_hh0, b_ih1, b_hh1,
                                                  wih0b, whh0b, wih1b, whh1b,
                                                  h0f, h1f, h0b, h1b,
                                                  out, h1fin, bar);
}
